// MOE_47459388621300
// MI455X (gfx1250) — compile-verified, matrix operand order checked by reading, not run on hardware
//
#include <hip/hip_runtime.h>
#include <hip/hip_bf16.h>

// ---------------------------------------------------------------------------
// MoE grouped SwiGLU FFN for MI455X (gfx1250), bf16 WMMA path with register
// software-pipelined global->LDS staging and batched fragment loads.
//   E=160 experts, CAP=120, S=3200, H=2048, I=192, K_TOP=6
// ---------------------------------------------------------------------------

#define NE     160
#define NCAP   120
#define NS     3200
#define NH     2048
#define NI     192
#define NKTOP  6

typedef __attribute__((ext_vector_type(16))) __bf16 v16bf;
typedef __attribute__((ext_vector_type(8)))  float  v8f;

// ----- helpers --------------------------------------------------------------

__device__ __forceinline__ unsigned pk_bf16(float a, float b) {
#if __has_builtin(__builtin_amdgcn_cvt_pk_bf16_f32)
    auto t = __builtin_amdgcn_cvt_pk_bf16_f32(a, b);
    unsigned r;
    __builtin_memcpy(&r, &t, sizeof(r));
    return r;
#else
    // software RNE + v_perm_b32 merge of the two high halves
    unsigned x = __float_as_uint(a);
    unsigned y = __float_as_uint(b);
    x += 0x7FFFu + ((x >> 16) & 1u);
    y += 0x7FFFu + ((y >> 16) & 1u);
    return __builtin_amdgcn_perm(y, x, 0x07060302u);   // {y.hi16, x.hi16}
#endif
}

__device__ __forceinline__ unsigned short bf16_bits(float a) {
    unsigned x = __float_as_uint(a);
    x += 0x7FFFu + ((x >> 16) & 1u);
    return (unsigned short)(x >> 16);
}

__device__ __forceinline__ uint4 pack8(float4 f0, float4 f1) {
    return make_uint4(pk_bf16(f0.x, f0.y), pk_bf16(f0.z, f0.w),
                      pk_bf16(f1.x, f1.y), pk_bf16(f1.z, f1.w));
}

union FragU { uint4 q[2]; v16bf v; };

// 16x32 bf16 fragment load (A-style layout; also used for B supplied as
// N x K row-major): lane r (mod 16) holds row r; lanes 0-15 hold K=[0,8)
// and [16,24); lanes 16-31 hold K=[8,16) and [24,32). Two ds_load_b128.
__device__ __forceinline__ v16bf load_frag(const unsigned short* buf, int strideElems,
                                           int row0, int k0, int lane) {
    int r  = row0 + (lane & 15);
    int kh = (lane >> 4) << 3;               // 0 or 8
    const unsigned short* p = buf + (size_t)r * strideElems + k0 + kh;
    FragU u;
    u.q[0] = *(const uint4*)(p);
    u.q[1] = *(const uint4*)(p + 16);
    return u.v;
}

#define WMMA_BF16(A, B, C) \
    __builtin_amdgcn_wmma_f32_16x16x32_bf16(false, (A), false, (B), (short)0, (C), false, false)

// ----- small prep kernels ---------------------------------------------------

__global__ void moe_inv_perm_kernel(const int* __restrict__ re_index,
                                    int* __restrict__ inv, int n) {
    int j = blockIdx.x * blockDim.x + threadIdx.x;
    if (j < n) inv[re_index[j]] = j;
}

__global__ void moe_init_out_kernel(const float* __restrict__ shared_out,
                                    float* __restrict__ out, int n) {
    int i = blockIdx.x * blockDim.x + threadIdx.x;
    if (i < n) out[i] = shared_out[i];
}

// ----- main expert kernel: one workgroup per expert, 512 threads = 16 waves -
//
// LDS layout (dynamic, ~104 KB); padded strides (40 / 200 elems) give
// conflict-free ds_load_b128 fragment reads.

#define LDS_A    0
#define LDS_BG   10240
#define LDS_BU   25600
#define LDS_BD   0
#define LDS_HB   51200
#define LDS_TIDX 102400
#define LDS_SARR 102912
#define LDS_WARR 103424
#define LDS_BYTES 103936

__global__ __launch_bounds__(512)
void moe_expert_kernel(const float* __restrict__ hidden,
                       const int*   __restrict__ token_index,
                       const float* __restrict__ topk_weight,
                       const float* __restrict__ gate_w,
                       const float* __restrict__ up_w,
                       const float* __restrict__ down_w,
                       const int*   __restrict__ inv,
                       float*       __restrict__ out) {
    extern __shared__ __align__(16) char smem[];
    unsigned short* Abuf = (unsigned short*)(smem + LDS_A);
    unsigned short* Bg   = (unsigned short*)(smem + LDS_BG);
    unsigned short* Bu   = (unsigned short*)(smem + LDS_BU);
    unsigned short* Bd   = (unsigned short*)(smem + LDS_BD);
    unsigned short* Hb   = (unsigned short*)(smem + LDS_HB);
    int*   tIdx = (int*)(smem + LDS_TIDX);
    int*   sArr = (int*)(smem + LDS_SARR);
    float* wArr = (float*)(smem + LDS_WARR);

    const int e    = blockIdx.x;
    const int tid  = threadIdx.x;
    const int w    = tid >> 5;        // wave 0..15
    const int lane = tid & 31;

    // Per-expert routing metadata.
    for (int c = tid; c < 128; c += 512) {
        if (c < NCAP) {
            tIdx[c] = token_index[e * NCAP + c];
            int j = inv[e * NCAP + c];
            int k = j / NS;
            int s = j - k * NS;
            sArr[c] = s;
            wArr[c] = topk_weight[k * NS + s];
        } else {
            tIdx[c] = -1;             // zero-padded rows 120..127
        }
    }
    __syncthreads();

    // ---------------- Phase 1: G = X Wg^T, U = X Wu^T  (M=128, N=192, K=2048)
    const int mg = w >> 2;            // M tiles {2mg, 2mg+1}
    const int ng = w & 3;             // N tiles {3ng .. 3ng+2}

    v8f accG[2][3] = {};
    v8f accU[2][3] = {};

    // Wave-uniform staging assignments (register software pipeline).
    const int aRow = tid >> 2;                  // 0..127
    const int aCol = (tid & 3) << 3;            // 0,8,16,24
    const int aTok = tIdx[aRow];                // hoisted token id

    float4 rA0, rA1;                            // A slice regs (8 floats)
    float4 rB0[3], rB1[3];                      // Wg/Wu pooled (24 floats)

    auto p1_issue = [&](int kb) {
        rA0 = make_float4(0.f, 0.f, 0.f, 0.f);
        rA1 = rA0;
        if (aTok >= 0) {
            const float* s = hidden + (size_t)aTok * NH + kb + aCol;
            rA0 = *(const float4*)(s);
            rA1 = *(const float4*)(s + 4);
        }
        #pragma unroll
        for (int j = 0; j < 3; ++j) {
            int u  = tid + 512 * j;             // 0..1535 ; <768 => Wg
            int uu = (u >= 768) ? (u - 768) : u;
            const float* base = (u >= 768) ? up_w : gate_w;
            int row = uu >> 2;
            int col = (uu & 3) << 3;
            const float* s = base + ((size_t)e * NI + row) * NH + kb + col;
            rB0[j] = *(const float4*)(s);
            rB1[j] = *(const float4*)(s + 4);
        }
    };
    auto p1_store = [&]() {
        *(uint4*)(Abuf + aRow * 40 + aCol) = pack8(rA0, rA1);
        #pragma unroll
        for (int j = 0; j < 3; ++j) {
            int u  = tid + 512 * j;
            int uu = (u >= 768) ? (u - 768) : u;
            unsigned short* dst = (u >= 768) ? Bu : Bg;
            int row = uu >> 2;
            int col = (uu & 3) << 3;
            *(uint4*)(dst + row * 40 + col) = pack8(rB0[j], rB1[j]);
        }
    };

    p1_issue(0);
    for (int kb = 0; kb < NH; kb += 32) {
        __syncthreads();              // staging buffers free to overwrite
        p1_store();                   // waits on in-flight global loads
        __syncthreads();              // slice visible
        if (kb + 32 < NH) p1_issue(kb + 32);   // overlap next loads w/ WMMAs

        // Batch ALL fragment loads first so the scheduler can clause the
        // ds_load_b128s and use partial s_wait_dscnt instead of 0-waits.
        v16bf aF[2], bgF[3], buF[3];
        aF[0] = load_frag(Abuf, 40, (mg * 2 + 0) * 16, 0, lane);
        aF[1] = load_frag(Abuf, 40, (mg * 2 + 1) * 16, 0, lane);
        #pragma unroll
        for (int ni = 0; ni < 3; ++ni) {
            int nt = ng * 3 + ni;
            bgF[ni] = load_frag(Bg, 40, nt * 16, 0, lane);
            buF[ni] = load_frag(Bu, 40, nt * 16, 0, lane);
        }
        #pragma unroll
        for (int ni = 0; ni < 3; ++ni) {
            accG[0][ni] = WMMA_BF16(aF[0], bgF[ni], accG[0][ni]);
            accG[1][ni] = WMMA_BF16(aF[1], bgF[ni], accG[1][ni]);
            accU[0][ni] = WMMA_BF16(aF[0], buF[ni], accU[0][ni]);
            accU[1][ni] = WMMA_BF16(aF[1], buF[ni], accU[1][ni]);
        }
    }

    // SwiGLU epilogue in registers -> resident bf16 H buffer.
    // C layout: vgpr v, lanes 0-15 -> (M=v, N=lane); lanes 16-31 -> (M=v+8).
    #pragma unroll
    for (int mi = 0; mi < 2; ++mi) {
        #pragma unroll
        for (int ni = 0; ni < 3; ++ni) {
            int row0 = (mg * 2 + mi) * 16 + (lane >> 4) * 8;
            int ncol = (ng * 3 + ni) * 16 + (lane & 15);
            #pragma unroll
            for (int v = 0; v < 8; ++v) {
                float gv = accG[mi][ni][v];
                float hv = (gv / (1.0f + __expf(-gv))) * accU[mi][ni][v];
                Hb[(row0 + v) * 200 + ncol] = bf16_bits(hv);
            }
        }
    }
    __syncthreads();

    // ---------------- Phase 2: O = H Wd^T  (M=128, N=2048, K=192)
    const int mg2 = w >> 2;
    const int ng2 = w & 3;

    // Chunk-invariant scatter metadata, hoisted to registers.
    int   sRowR[2][8];
    float wRowR[2][8];
    #pragma unroll
    for (int mi = 0; mi < 2; ++mi) {
        #pragma unroll
        for (int v = 0; v < 8; ++v) {
            int c = (mg2 * 2 + mi) * 16 + (lane >> 4) * 8 + v;
            bool ok = c < NCAP;
            sRowR[mi][v] = ok ? sArr[c] : -1;
            wRowR[mi][v] = ok ? wArr[c] : 0.0f;
        }
    }

    float4 rD0[6], rD1[6];            // 48 floats of Wd staging
    auto p2_issue = [&](int nc) {
        #pragma unroll
        for (int j = 0; j < 6; ++j) {
            int u   = tid + 512 * j;          // 0..3071
            int row = u / 24;                 // 0..127
            int col = (u - row * 24) << 3;    // 0..184
            const float* s = down_w + ((size_t)e * NH + nc + row) * NI + col;
            rD0[j] = *(const float4*)(s);
            rD1[j] = *(const float4*)(s + 4);
        }
    };
    auto p2_store = [&]() {
        #pragma unroll
        for (int j = 0; j < 6; ++j) {
            int u   = tid + 512 * j;
            int row = u / 24;
            int col = (u - row * 24) << 3;
            *(uint4*)(Bd + row * 200 + col) = pack8(rD0[j], rD1[j]);
        }
    };

    p2_issue(0);
    for (int nc = 0; nc < NH; nc += 128) {
        __syncthreads();              // Bd region free
        p2_store();
        __syncthreads();
        if (nc + 128 < NH) p2_issue(nc + 128);

        v8f acc[2][2] = {};
        #pragma unroll
        for (int kk = 0; kk < NI; kk += 32) {
            v16bf aF0 = load_frag(Hb, 200, (mg2 * 2 + 0) * 16, kk, lane);
            v16bf aF1 = load_frag(Hb, 200, (mg2 * 2 + 1) * 16, kk, lane);
            v16bf bF0 = load_frag(Bd, 200, (ng2 * 2 + 0) * 16, kk, lane);
            v16bf bF1 = load_frag(Bd, 200, (ng2 * 2 + 1) * 16, kk, lane);
            acc[0][0] = WMMA_BF16(aF0, bF0, acc[0][0]);
            acc[0][1] = WMMA_BF16(aF0, bF1, acc[0][1]);
            acc[1][0] = WMMA_BF16(aF1, bF0, acc[1][0]);
            acc[1][1] = WMMA_BF16(aF1, bF1, acc[1][1]);
        }

        // Weighted scatter into out[s, h]; padded rows have sRow = -1.
        #pragma unroll
        for (int mi = 0; mi < 2; ++mi) {
            #pragma unroll
            for (int ni = 0; ni < 2; ++ni) {
                int hcol = nc + (ng2 * 2 + ni) * 16 + (lane & 15);
                #pragma unroll
                for (int v = 0; v < 8; ++v) {
                    if (sRowR[mi][v] >= 0) {
                        atomicAdd(out + (size_t)sRowR[mi][v] * NH + hcol,
                                  acc[mi][ni][v] * wRowR[mi][v]);
                    }
                }
            }
        }
    }
}

// ---------------------------------------------------------------------------

extern "C" void kernel_launch(void* const* d_in, const int* in_sizes, int n_in,
                              void* d_out, int out_size, void* d_ws, size_t ws_size,
                              hipStream_t stream) {
    const float* hidden      = (const float*)d_in[0];
    const int*   token_index = (const int*)  d_in[1];
    const int*   re_index    = (const int*)  d_in[2];
    const float* topk_weight = (const float*)d_in[3];
    const float* shared_out  = (const float*)d_in[4];
    const float* gate_w      = (const float*)d_in[5];
    const float* up_w        = (const float*)d_in[6];
    const float* down_w      = (const float*)d_in[7];
    float* out = (float*)d_out;
    int*   inv = (int*)d_ws;                 // E*CAP ints of scratch

    {
        int n = NE * NCAP;
        moe_inv_perm_kernel<<<(n + 255) / 256, 256, 0, stream>>>(re_index, inv, n);
    }
    {
        int n = NS * NH;
        moe_init_out_kernel<<<(n + 255) / 256, 256, 0, stream>>>(shared_out, out, n);
    }
    moe_expert_kernel<<<NE, 512, LDS_BYTES, stream>>>(
        hidden, token_index, topk_weight, gate_w, up_w, down_w, inv, out);
}
